// BasicBlockShared_70222715289971
// MI455X (gfx1250) — compile-verified, matrix operand order checked by reading, not run on hardware
//
#include <hip/hip_runtime.h>

// ---------------------------------------------------------------------------
// Problem constants (match reference)
// ---------------------------------------------------------------------------
#define BB   32      // batch
#define CC   256     // channels
#define HH   32
#define WW   32
#define EE   8       // experts
#define HWN  (HH*WW)             // 1024 spatial positions (GEMM N)
#define EPSV 1e-5f

// Async global->LDS path (gfx1250), guarded so compilation can't regress.
#if defined(__has_builtin)
#  if __has_builtin(__builtin_amdgcn_global_load_async_to_lds_b128) && \
      __has_builtin(__builtin_amdgcn_s_wait_asynccnt)
#    define USE_ASYNC_LDS 1
#  endif
#endif
#ifndef USE_ASYNC_LDS
#  define USE_ASYNC_LDS 0
#endif

// WMMA vector types (gfx1250, wave32)
typedef __attribute__((ext_vector_type(16))) __bf16 v16bf;
typedef __attribute__((ext_vector_type(8)))  float  v8f;
typedef __attribute__((ext_vector_type(4)))  int    v4i;

#define GLOBAL_AS __attribute__((address_space(1)))
#define LDS_AS    __attribute__((address_space(3)))

// bf16 helpers (storage-only: keep data as ushort, convert via bit ops)
__device__ __forceinline__ unsigned short f2bf(float f) {
    unsigned u = __float_as_uint(f);
    u += 0x7FFFu + ((u >> 16) & 1u);      // round-to-nearest-even
    return (unsigned short)(u >> 16);
}

// ---------------------------------------------------------------------------
// 1) Router: GAP -> Linear(C,E) -> sigmoid.  One block per batch sample.
// ---------------------------------------------------------------------------
__global__ __launch_bounds__(256)
void router_kernel(const float* __restrict__ x,
                   const float* __restrict__ rw,   // [E][C]
                   const float* __restrict__ rb,   // [E]
                   float* __restrict__ r)          // [B][E]
{
    const int b    = blockIdx.x;
    const int tid  = threadIdx.x;
    const int lane = tid & 31;
    const int wv   = tid >> 5;         // 8 waves
    __shared__ float pooled[CC];

    for (int cb = 0; cb < CC; cb += 8) {
        const int c = cb + wv;
        const float* p = x + ((long)b * CC + c) * HWN;
        float s = 0.f;
        for (int j = lane; j < HWN; j += 32) s += p[j];   // coalesced
        for (int off = 16; off; off >>= 1) s += __shfl_down(s, off, 32);
        if (lane == 0) pooled[c] = s * (1.0f / (float)HWN);
    }
    __syncthreads();
    if (tid < EE) {
        float s = rb[tid];
        const float* wrow = rw + tid * CC;
        for (int c = 0; c < CC; ++c) s += pooled[c] * wrow[c];
        r[b * EE + tid] = 1.0f / (1.0f + __expf(-s));
    }
}

// ---------------------------------------------------------------------------
// 2) BN eval-mode scale/shift precompute for both layers.
// ---------------------------------------------------------------------------
__global__ __launch_bounds__(256)
void prep_bn_kernel(const float* g1, const float* b1, const float* m1, const float* v1,
                    const float* g2, const float* b2, const float* m2, const float* v2,
                    float* s1, float* t1, float* s2, float* t2)
{
    const int c = threadIdx.x;
    float sc1 = g1[c] * rsqrtf(v1[c] + EPSV);
    s1[c] = sc1;  t1[c] = b1[c] - m1[c] * sc1;
    float sc2 = g2[c] * rsqrtf(v2[c] + EPSV);
    s2[c] = sc2;  t2[c] = b2[c] - m2[c] * sc2;
}

// ---------------------------------------------------------------------------
// 3) fp32 -> bf16 convert (activations)
// ---------------------------------------------------------------------------
__global__ __launch_bounds__(256)
void convert_bf16_kernel(const float* __restrict__ in,
                         unsigned short* __restrict__ out, int n)
{
    int i = blockIdx.x * 256 + threadIdx.x;
    int stride = gridDim.x * 256;
    for (; i < n; i += stride) out[i] = f2bf(in[i]);
}

// ---------------------------------------------------------------------------
// 4) Combine expert kernels per sample, reorder tap-major, emit bf16.
//    wbank: [E][O][I][3][3] fp32  ->  wc: [B][9][O][I] bf16
// ---------------------------------------------------------------------------
__global__ __launch_bounds__(256)
void combine_weights_kernel(const float* __restrict__ wbank,
                            const float* __restrict__ r,    // [B][E]
                            unsigned short* __restrict__ wc)
{
    const int idx = blockIdx.x * 256 + threadIdx.x;        // B*O*I = 2,097,152
    const int i = idx & (CC - 1);
    const int o = (idx >> 8) & (CC - 1);
    const int b = idx >> 16;

    float rv[EE];
#pragma unroll
    for (int e = 0; e < EE; ++e) rv[e] = r[b * EE + e];

    float acc[9] = {0,0,0,0,0,0,0,0,0};
#pragma unroll
    for (int e = 0; e < EE; ++e) {
        const float* src = wbank + ((long)((e * CC + o) * CC + i)) * 9;
#pragma unroll
        for (int t = 0; t < 9; ++t) acc[t] += rv[e] * src[t];
    }
#pragma unroll
    for (int t = 0; t < 9; ++t)
        wc[(((long)b * 9 + t) * CC + o) * CC + i] = f2bf(acc[t]);
}

// ---------------------------------------------------------------------------
// 5) Implicit-GEMM 3x3 conv, tap decomposition, bf16 WMMA, f32 accum.
//    Block tile: 128 (M=o) x 64 (N=hw), 8 waves; wave = 32x32 (2x2 WMMA).
//    Software-pipelined: outer runtime loop over 8 channel chunks, inner
//    FULLY-UNROLLED loop over 9 taps (compile-time tap shifts, A buffer
//    parity = (ck&1)^(tap&1)).  Async global->LDS loads for tap+1 issued
//    right after the barrier, overlapping the WMMAs.  Haloed activation
//    tile (4x34x32) double-buffered, staged once per chunk at tap 8.
//    One barrier per tap.
//    mode 0: BN+ReLU -> bf16;  mode 1: BN + residual + ReLU -> f32
// ---------------------------------------------------------------------------
#define A_STRIDE 40            // halfs per A row (80B = 20 banks, conflict-free)
#define H_SLOT   40            // halfs per halo (hl,wl) slot
#define H_ELEMS  (4 * 34 * H_SLOT)

__global__ __launch_bounds__(256)
void condconv_wmma_kernel(const unsigned short* __restrict__ in,   // [B][C][H][W] bf16
                          const unsigned short* __restrict__ wc,   // [B][9][O][I] bf16
                          const float* __restrict__ sA,            // BN scale [C]
                          const float* __restrict__ tA,            // BN shift [C]
                          const float* __restrict__ xres,          // residual (mode 1)
                          unsigned short* __restrict__ out_bf,     // mode 0 output
                          float* __restrict__ out_f,               // mode 1 output
                          int mode)
{
    __shared__ __align__(16) unsigned short As[2][128][A_STRIDE];
    __shared__ __align__(16) unsigned short Hs[2][H_ELEMS];

    const int tid  = threadIdx.x;
    const int lane = tid & 31;
    const int wv   = tid >> 5;           // 0..7
    const int wm   = (wv >> 1) * 32;     // wave row offset within 128-tile
    const int wn   = (wv & 1) * 32;      // wave col offset within 64-tile
    const int hi   = lane >> 4;          // half-wave id
    const int lr   = lane & 15;

    const int bz  = blockIdx.z;
    const int om0 = blockIdx.y * 128;    // output-channel tile base
    const int n0  = blockIdx.x * 64;     // spatial tile base (2 image rows)
    const int h0  = n0 >> 5;

    const unsigned short* inb  = in + (long)bz * CC * HWN;
    const unsigned short* wcb0 = wc + (long)bz * 9 * (CC * CC);

    // Per-wave invariant LDS offsets (in halfs)
    int aoff[2];                          // A fragment row bases (chunk0 of pair)
#pragma unroll
    for (int mi = 0; mi < 2; ++mi)
        aoff[mi] = (wm + mi * 16 + lr) * A_STRIDE + hi * 8;
    int boff[2];                          // B fragment halo bases at (dh,dw)=(0,0)
#pragma unroll
    for (int ni = 0; ni < 2; ++ni) {
        const int n = wn + ni * 16 + lr;  // 0..63
        boff[ni] = (((n >> 5) + 1) * 34 + (n & 31) + 1) * H_SLOT + hi * 8;
    }
    // A-staging decomposition for this thread (2 segments of 16B)
    const int mrow0 = tid >> 2, kseg0 = tid & 3;         // seg = tid
    const int mrow1 = (tid + 256) >> 2, kseg1 = tid & 3; // seg = tid + 256

    v8f acc[2][2];
#pragma unroll
    for (int mi = 0; mi < 2; ++mi)
#pragma unroll
        for (int ni = 0; ni < 2; ++ni)
            acc[mi][ni] = (v8f){0.f,0.f,0.f,0.f,0.f,0.f,0.f,0.f};

    // ---- staging helpers ----
    auto issueA = [&](int buf, int kc, int tap) {
        const unsigned short* wcb = wcb0 + tap * (CC * CC) + kc;
#if USE_ASYNC_LDS
        __builtin_amdgcn_global_load_async_to_lds_b128(
            (GLOBAL_AS v4i*)(void*)(wcb + (om0 + mrow0) * CC + kseg0 * 8),
            (LDS_AS v4i*)(void*)&As[buf][mrow0][kseg0 * 8], 0, 0);
        __builtin_amdgcn_global_load_async_to_lds_b128(
            (GLOBAL_AS v4i*)(void*)(wcb + (om0 + mrow1) * CC + kseg1 * 8),
            (LDS_AS v4i*)(void*)&As[buf][mrow1][kseg1 * 8], 0, 0);
#else
        *(uint4*)&As[buf][mrow0][kseg0 * 8] =
            *(const uint4*)(wcb + (om0 + mrow0) * CC + kseg0 * 8);
        *(uint4*)&As[buf][mrow1][kseg1 * 8] =
            *(const uint4*)(wcb + (om0 + mrow1) * CC + kseg1 * 8);
#endif
    };
    auto stageHalo = [&](int buf, int kc) {
        // 4352 entries; idx = ((hl*32)+k)*34 + wl  (wl fastest -> coalesced w)
#pragma unroll
        for (int it2 = 0; it2 < 17; ++it2) {
            const int idx = tid + it2 * 256;
            if (idx < 4 * 32 * 34) {
                const int wl = idx % 34;
                const int t2 = idx / 34;          // 0..127
                const int k  = t2 & 31;
                const int hl = t2 >> 5;           // 0..3
                const int hh = h0 - 1 + hl;
                const int ww = wl - 1;
                unsigned short v = 0;
                if (hh >= 0 && hh < HH && ww >= 0 && ww < WW)
                    v = inb[(kc + k) * HWN + hh * WW + ww];
                Hs[buf][(hl * 34 + wl) * H_SLOT + k] = v;
            }
        }
    };

    // ---- pipeline prologue ----
    stageHalo(0, 0);          // visible after first barrier
    issueA(0, 0, 0);

    // ---- main pipeline: 8 channel chunks (runtime) x 9 taps (unrolled) ----
#pragma unroll 1
    for (int ck = 0; ck < 8; ++ck) {
        const int ckpar = ck & 1;
        const int kc = ck * 32;
        const unsigned short* Hb = &Hs[ckpar][0];

#pragma unroll
        for (int tap = 0; tap < 9; ++tap) {
            const int cur = ckpar ^ (tap & 1);    // tap parity compile-time

#if USE_ASYNC_LDS
            __builtin_amdgcn_s_wait_asynccnt(0);  // own A loads for this tap done
#endif
            __syncthreads();                      // data ready; prev bufs free

            // prefetch next tap's A tile (and next chunk's halo) -> overlap WMMA
            if (tap < 8) {
                issueA(cur ^ 1, kc, tap + 1);
            } else if (ck < 7) {
                issueA(cur ^ 1, kc + 32, 0);
                stageHalo(ckpar ^ 1, kc + 32);
            }

            // ---- fragments + WMMA (compile-time tap shift) ----
            const int dh = tap / 3 - 1;
            const int dw = tap % 3 - 1;
            const int tshift = (dh * 34 + dw) * H_SLOT;
            const unsigned short* Ab = &As[cur][0][0];

            v16bf af[2];
#pragma unroll
            for (int mi = 0; mi < 2; ++mi) {
                ((uint4*)&af[mi])[0] = *(const uint4*)(Ab + aoff[mi]);
                ((uint4*)&af[mi])[1] = *(const uint4*)(Ab + aoff[mi] + 16);
            }
#pragma unroll
            for (int ni = 0; ni < 2; ++ni) {
                const unsigned short* bp = Hb + boff[ni] + tshift;
                v16bf bfr;
                ((uint4*)&bfr)[0] = *(const uint4*)(bp);
                ((uint4*)&bfr)[1] = *(const uint4*)(bp + 16);
#pragma unroll
                for (int mi = 0; mi < 2; ++mi) {
                    acc[mi][ni] = __builtin_amdgcn_wmma_f32_16x16x32_bf16(
                        false, af[mi], false, bfr, (short)0, acc[mi][ni],
                        false, false);
                }
            }
        }
    }

    // ---- epilogue: BN (+residual) + ReLU, coalesced along w ----
#pragma unroll
    for (int mi = 0; mi < 2; ++mi) {
#pragma unroll
        for (int ni = 0; ni < 2; ++ni) {
            const int ng = n0 + wn + ni * 16 + lr;          // global n in [0,1024)
            const int h = ng >> 5, w = ng & 31;
#pragma unroll
            for (int rr = 0; rr < 8; ++rr) {
                const int o = om0 + wm + mi * 16 + rr + hi * 8;
                float y = acc[mi][ni][rr] * sA[o] + tA[o];
                const long off = (((long)bz * CC + o) * HH + h) * WW + w;
                if (mode == 0) {
                    y = fmaxf(y, 0.f);
                    out_bf[off] = f2bf(y);
                } else {
                    y += xres[off];
                    out_f[off] = fmaxf(y, 0.f);
                }
            }
        }
    }
}

// ---------------------------------------------------------------------------
// Launch
// ---------------------------------------------------------------------------
extern "C" void kernel_launch(void* const* d_in, const int* in_sizes, int n_in,
                              void* d_out, int out_size, void* d_ws, size_t ws_size,
                              hipStream_t stream) {
    const float* x    = (const float*)d_in[0];
    const float* rw   = (const float*)d_in[1];
    const float* rb   = (const float*)d_in[2];
    const float* w1   = (const float*)d_in[3];
    const float* w2   = (const float*)d_in[4];
    const float* g1   = (const float*)d_in[5];
    const float* b1   = (const float*)d_in[6];
    const float* m1   = (const float*)d_in[7];
    const float* v1   = (const float*)d_in[8];
    const float* g2   = (const float*)d_in[9];
    const float* b2   = (const float*)d_in[10];
    const float* m2   = (const float*)d_in[11];
    const float* v2   = (const float*)d_in[12];
    float* out = (float*)d_out;

    // workspace carve (all 16B aligned)
    char* ws = (char*)d_ws;
    float* r_ws = (float*)ws;                 // B*E = 256 floats
    float* s1   = r_ws + 256;
    float* t1   = s1 + CC;
    float* t2s  = t1 + CC;   // s2
    float* t2t  = t2s + CC;  // t2
    unsigned short* xbf   = (unsigned short*)(ws + 8192);
    const long NX = (long)BB * CC * HWN;                       // 8,388,608
    unsigned short* midbf = xbf + NX;
    unsigned short* wcomb = midbf + NX;                        // B*9*C*C bf16

    // 1) router
    router_kernel<<<BB, 256, 0, stream>>>(x, rw, rb, r_ws);
    // 2) BN prep
    prep_bn_kernel<<<1, 256, 0, stream>>>(g1, b1, m1, v1, g2, b2, m2, v2,
                                          s1, t1, t2s, t2t);
    // 3) x -> bf16
    convert_bf16_kernel<<<2048, 256, 0, stream>>>(x, xbf, (int)NX);

    const dim3 cgrid(HWN / 64, CC / 128, BB);   // (16, 2, 32)

    // 4) combine w1 -> conv1 (BN1+ReLU -> bf16)
    combine_weights_kernel<<<(BB * CC * CC) / 256, 256, 0, stream>>>(w1, r_ws, wcomb);
    condconv_wmma_kernel<<<cgrid, 256, 0, stream>>>(
        xbf, wcomb, s1, t1, nullptr, midbf, nullptr, 0);

    // 5) combine w2 (reuse buffer) -> conv2 (BN2 + residual + ReLU -> f32)
    combine_weights_kernel<<<(BB * CC * CC) / 256, 256, 0, stream>>>(w2, r_ws, wcomb);
    condconv_wmma_kernel<<<cgrid, 256, 0, stream>>>(
        midbf, wcomb, t2s, t2t, x, nullptr, out, 1);
}
